// CustomAttention_54494545052068
// MI455X (gfx1250) — matrix operand order repaired and checked by reading, not compiled, not run on hardware
//
#include <hip/hip_runtime.h>
#include <hip/hip_bf16.h>

// ---------------- types ----------------
typedef __attribute__((ext_vector_type(16))) __bf16 v16bf;
typedef __attribute__((ext_vector_type(8)))  __bf16 v8bf;
typedef __attribute__((ext_vector_type(8)))  float  v8f;

#define Hh   16
#define Sq   2048
#define HID  2048
#define DQK  192   // DN + DR
#define DN   128
#define DR   64
#define DV   128
#define NQ   3072  // H*(DN+DR)
#define NV   2048  // H*DV

static __device__ __forceinline__ unsigned short f32_to_bf16(float f) {
    unsigned int u = __float_as_uint(f);
    unsigned int lsb = (u >> 16) & 1u;
    u += 0x7fffu + lsb;
    return (unsigned short)(u >> 16);
}
static __device__ __forceinline__ float bf16_to_f32(unsigned short h) {
    return __uint_as_float(((unsigned int)h) << 16);
}

static __device__ __forceinline__ v8f wmma_bf16(v16bf a, v16bf b, v8f c) {
    return __builtin_amdgcn_wmma_f32_16x16x32_bf16(false, a, false, b, (short)0, c,
                                                   false, false);
}
static __device__ __forceinline__ v16bf cat8(v8bf lo, v8bf hi) {
    return __builtin_shufflevector(lo, hi, 0,1,2,3,4,5,6,7,8,9,10,11,12,13,14,15);
}
static __device__ __forceinline__ v8f vzero8() {
    v8f z;
#pragma unroll
    for (int i = 0; i < 8; ++i) z[i] = 0.0f;
    return z;
}

// Async copy of one 16B chunk: global -> LDS, tracked by ASYNCcnt.
static __device__ __forceinline__ void async_b128(const unsigned short* gp,
                                                  const unsigned short* lp) {
    const unsigned long long ga = (unsigned long long)(uintptr_t)gp;
    const unsigned int       la = (unsigned int)(uintptr_t)lp;  // low 32 = LDS offset
    asm volatile("global_load_async_to_lds_b128 %0, %1, off"
                 :: "v"(la), "v"(ga) : "memory");
}
static __device__ __forceinline__ void wait_async0() {
    asm volatile("s_wait_asynccnt 0" ::: "memory");
}

// ---------------- f32 -> bf16 convert ----------------
__global__ void conv_f32_bf16(const float* __restrict__ in,
                              unsigned short* __restrict__ out, int n) {
    int i = blockIdx.x * blockDim.x + threadIdx.x;
    if (i < n) out[i] = f32_to_bf16(in[i]);
}

// ---------------- NT GEMM: C = A(MxK) * B(NxK)^T, LDS double-buffered ------
// mode 0: outF[m*N+n] = c                          (f32, final projection)
// mode 1: packed per-head QK layout [H][S][192]    (bf16)
// mode 2: transposed V layout: outB[n*M + m]       (bf16, Vt = [H*DV][S])
#define GAROW 40  // LDS row stride (elems): 32 data + 8 pad = 80B (bank-spread)
__global__ __launch_bounds__(256)
void gemm_nt_bf16(const unsigned short* __restrict__ A,
                  const unsigned short* __restrict__ B,
                  float* __restrict__ outF,
                  unsigned short* __restrict__ outB,
                  int M, int N, int K, int mode) {
    __shared__ __align__(16) unsigned short Abuf[2][128 * GAROW];  // 2 x 10 KB
    __shared__ __align__(16) unsigned short Bbuf[2][64 * GAROW];   // 2 x 5 KB

    const int tid  = threadIdx.x;
    const int wave = tid >> 5;
    const int lane = tid & 31;
    const int lam  = lane & 15;
    const int hh   = lane >> 4;
    const int wm   = wave >> 1;     // 0..3
    const int wn   = wave & 1;      // 0..1
    const int mblk = blockIdx.x * 128;
    const int nblk = blockIdx.y * 64;

    v8f c[2][2];
#pragma unroll
    for (int i = 0; i < 2; ++i)
#pragma unroll
        for (int j = 0; j < 2; ++j) c[i][j] = vzero8();

    // --- async stage of one 32-deep K slab into LDS buffer bb ---
    auto stage = [&](int bb, int k0) {
        // A tile: 128 rows x 32 elems = 512 x 16B chunks -> 2 per thread
#pragma unroll
        for (int e = 0; e < 2; ++e) {
            const int ch  = tid * 2 + e;
            const int row = ch >> 2;
            const int col = (ch & 3) * 8;
            async_b128(A + (size_t)(mblk + row) * K + k0 + col,
                       &Abuf[bb][row * GAROW + col]);
        }
        // B tile: 64 rows x 32 elems = 256 chunks -> 1 per thread
        {
            const int row = tid >> 2;
            const int col = (tid & 3) * 8;
            async_b128(B + (size_t)(nblk + row) * K + k0 + col,
                       &Bbuf[bb][row * GAROW + col]);
        }
        if (k0 + 32 < K)  // keep the slab after next warm in L2
            __builtin_prefetch(A + (size_t)(mblk + lam) * K + k0 + 32, 0, 1);
    };

    stage(0, 0);
    int bb = 0;
    for (int k0 = 0; k0 < K; k0 += 32) {
        wait_async0();
        __syncthreads();
        if (k0 + 32 < K) stage(bb ^ 1, k0 + 32);

        v16bf af[2], bf[2];
#pragma unroll
        for (int i = 0; i < 2; ++i) {
            const unsigned short* ar = &Abuf[bb][(wm * 32 + i * 16 + lam) * GAROW];
            af[i] = cat8(*(const v8bf*)(ar + hh * 16),
                         *(const v8bf*)(ar + hh * 16 + 8));
            const unsigned short* br =
                &Bbuf[bb][(wn * 32 + i * 16 + lam) * GAROW + hh * 16];
            bf[i] = *(const v16bf*)br;
        }
#pragma unroll
        for (int i = 0; i < 2; ++i)
#pragma unroll
            for (int j = 0; j < 2; ++j)
                c[i][j] = wmma_bf16(af[i], bf[j], c[i][j]);
        bb ^= 1;
    }

    // store: C layout -> VGPR r holds row (r + 8*hh), column lam
#pragma unroll
    for (int i = 0; i < 2; ++i) {
#pragma unroll
        for (int j = 0; j < 2; ++j) {
#pragma unroll
            for (int r = 0; r < 8; ++r) {
                const int row = mblk + wm * 32 + i * 16 + r + 8 * hh;
                const int col = nblk + wn * 32 + j * 16 + lam;
                const float val = c[i][j][r];
                if (mode == 0) {
                    outF[(size_t)row * N + col] = val;
                } else if (mode == 1) {
                    size_t idx;
                    if (col < Hh * DN) {
                        const int hd = col >> 7, d = col & 127;
                        idx = ((size_t)hd * Sq + row) * DQK + d;
                    } else {
                        const int cc = col - Hh * DN;
                        const int hd = cc >> 6, d = cc & 63;
                        idx = ((size_t)hd * Sq + row) * DQK + DN + d;
                    }
                    outB[idx] = f32_to_bf16(val);
                } else {  // mode 2: Vt[n][m]
                    outB[(size_t)col * M + row] = f32_to_bf16(val);
                }
            }
        }
    }
}

// ---------------- RoPE (+ score scaling), in place on [H][S][192] bf16 -----
__global__ __launch_bounds__(256)
void rope_scale(unsigned short* __restrict__ buf, float scaleN, float scaleR) {
    const int row  = blockIdx.x * 8 + (threadIdx.x >> 5);  // over H*S
    const int j    = threadIdx.x & 31;                     // rope pair index
    if (row >= Hh * Sq) return;
    const int s = row & (Sq - 1);
    unsigned short* r = buf + (size_t)row * DQK;

    const float ang = (float)s * __powf(10000.0f, -(float)j * (1.0f / 32.0f));
    const float cs = __cosf(ang), sn = __sinf(ang);
    const float x1 = bf16_to_f32(r[DN + j]);
    const float x2 = bf16_to_f32(r[DN + 32 + j]);
    r[DN + j]      = f32_to_bf16((x1 * cs - x2 * sn) * scaleR);
    r[DN + 32 + j] = f32_to_bf16((x1 * sn + x2 * cs) * scaleR);

    if (scaleN != 1.0f) {
#pragma unroll
        for (int t = 0; t < 4; ++t) {
            const int d = j + t * 32;
            r[d] = f32_to_bf16(bf16_to_f32(r[d]) * scaleN);
        }
    }
}

// ---------------- flash attention, K/V staged to LDS via async copy --------
#define KROW 200  // 192 + 8 pad elems -> 400B row stride (bank-spread)
#define VROW 40   // 32 + 8 pad elems  ->  80B row stride
__global__ __launch_bounds__(256)
void flash_attn(const unsigned short* __restrict__ Qp,   // [H][S][192] (pre-scaled)
                const unsigned short* __restrict__ Kp,   // [H][S][192]
                const unsigned short* __restrict__ Vt,   // [H][128][S]
                unsigned short* __restrict__ ctx) {      // [S][H*128]
    __shared__ __align__(16) unsigned short Kbuf[2][32 * KROW];   // 2 x 12.5 KB
    __shared__ __align__(16) unsigned short Vbuf[2][128 * VROW];  // 2 x 10 KB
    __shared__ __align__(16) unsigned short pbuf[8][16 * 32];     // per-wave P

    const int tid  = threadIdx.x;
    const int wave = tid >> 5;
    const int lane = tid & 31;
    const int lam  = lane & 15;
    const int hh   = lane >> 4;
    const int h    = blockIdx.x;
    const int q0   = blockIdx.y * 128 + wave * 16;

    const unsigned short* Qh = Qp + (size_t)h * Sq * DQK;
    const unsigned short* Kh = Kp + (size_t)h * Sq * DQK;
    const unsigned short* Vh = Vt + (size_t)h * DV * Sq;

    // Q fragments: 16 x 192 = 6 A-frags of 16x32 (registers, reused all loop)
    v16bf qf[6];
#pragma unroll
    for (int kc = 0; kc < 6; ++kc) {
        const unsigned short* base = Qh + (size_t)(q0 + lam) * DQK + kc * 32;
        qf[kc] = cat8(*(const v8bf*)(base + hh * 16),
                      *(const v8bf*)(base + hh * 16 + 8));
    }

    v8f o[8];
#pragma unroll
    for (int t = 0; t < 8; ++t) o[t] = vzero8();
    float mrun[8], lrun[8];
#pragma unroll
    for (int j = 0; j < 8; ++j) { mrun[j] = -3.0e38f; lrun[j] = 0.0f; }

    unsigned short* pw = &pbuf[wave][0];

    // --- async stage of one 32-key K/V chunk (shared by all 8 waves) ---
    auto stage_kv = [&](int bb, int kb) {
        // K chunk: 32 rows x 192 elems = 768 x 16B chunks -> 3 per thread
#pragma unroll
        for (int e = 0; e < 3; ++e) {
            const int ch  = tid * 3 + e;
            const int row = ch / 24;
            const int col = (ch % 24) * 8;
            async_b128(Kh + (size_t)(kb + row) * DQK + col,
                       &Kbuf[bb][row * KROW + col]);
        }
        // V chunk: 128 rows x 32 elems = 512 chunks -> 2 per thread
#pragma unroll
        for (int e = 0; e < 2; ++e) {
            const int ch  = tid * 2 + e;
            const int row = ch >> 2;
            const int col = (ch & 3) * 8;
            async_b128(Vh + (size_t)row * Sq + kb + col,
                       &Vbuf[bb][row * VROW + col]);
        }
    };

    stage_kv(0, 0);
    int bb = 0;
    for (int kb = 0; kb < Sq; kb += 32) {
        wait_async0();
        __syncthreads();
        if (kb + 32 < Sq) stage_kv(bb ^ 1, kb + 32);

        // ---- scores: two 16x16 tiles over 192-deep dot product (K from LDS)
        v8f c0 = vzero8(), c1 = vzero8();
#pragma unroll
        for (int kc = 0; kc < 6; ++kc) {
            const v16bf b0 =
                *(const v16bf*)(&Kbuf[bb][lam * KROW + kc * 32 + hh * 16]);
            const v16bf b1 =
                *(const v16bf*)(&Kbuf[bb][(16 + lam) * KROW + kc * 32 + hh * 16]);
            c0 = wmma_bf16(qf[kc], b0, c0);
            c1 = wmma_bf16(qf[kc], b1, c1);
        }

        // ---- online softmax (row r = j + 8*hh lives in one 16-lane half)
#pragma unroll
        for (int j = 0; j < 8; ++j) {
            float m = fmaxf(c0[j], c1[j]);
            for (int msk = 1; msk < 16; msk <<= 1)
                m = fmaxf(m, __shfl_xor(m, msk, 32));
            const float mnew = fmaxf(mrun[j], m);
            const float corr = __expf(mrun[j] - mnew);
            const float p0 = __expf(c0[j] - mnew);
            const float p1 = __expf(c1[j] - mnew);
            float rs = p0 + p1;
            for (int msk = 1; msk < 16; msk <<= 1)
                rs += __shfl_xor(rs, msk, 32);
            lrun[j] = lrun[j] * corr + rs;
            mrun[j] = mnew;
#pragma unroll
            for (int t = 0; t < 8; ++t) o[t][j] *= corr;
            const int r = j + 8 * hh;
            pw[r * 32 + lam]      = f32_to_bf16(p0);
            pw[r * 32 + 16 + lam] = f32_to_bf16(p1);
        }
        asm volatile("s_wait_dscnt 0" ::: "memory");  // P visible across lanes

        // ---- P (16x32) x V (32x128), V from LDS
        const unsigned short* pr = pw + lam * 32;
        const v16bf pf = cat8(*(const v8bf*)(pr + hh * 16),
                              *(const v8bf*)(pr + hh * 16 + 8));
#pragma unroll
        for (int nt = 0; nt < 8; ++nt) {
            const v16bf vb =
                *(const v16bf*)(&Vbuf[bb][(nt * 16 + lam) * VROW + hh * 16]);
            o[nt] = wmma_bf16(pf, vb, o[nt]);
        }
        bb ^= 1;
    }

    // ---- epilogue: normalize rows and emit ctx[s][h*128 + d] in bf16
#pragma unroll
    for (int j = 0; j < 8; ++j) {
        const float rinv = 1.0f / lrun[j];
        const int row = q0 + j + 8 * hh;
#pragma unroll
        for (int nt = 0; nt < 8; ++nt) {
            ctx[(size_t)row * NV + h * DV + nt * 16 + lam] =
                f32_to_bf16(o[nt][j] * rinv);
        }
    }
}

// ---------------- host side ----------------
extern "C" void kernel_launch(void* const* d_in, const int* in_sizes, int n_in,
                              void* d_out, int out_size, void* d_ws, size_t ws_size,
                              hipStream_t stream) {
    const float* X  = (const float*)d_in[0];  // [1,2048,2048]
    const float* Wq = (const float*)d_in[1];  // [3072,2048]
    const float* Wk = (const float*)d_in[2];  // [3072,2048]
    const float* Wv = (const float*)d_in[3];  // [2048,2048]
    const float* Wo = (const float*)d_in[4];  // [2048,2048]
    float* out = (float*)d_out;               // [2048,2048]

    const size_t nX  = (size_t)Sq * HID;
    const size_t nWq = (size_t)NQ * HID;
    const size_t nWv = (size_t)NV * HID;
    const size_t nQp = (size_t)Hh * Sq * DQK;
    const size_t nVt = (size_t)NV * Sq;
    const size_t nCx = (size_t)Sq * NV;

    unsigned short* p   = (unsigned short*)d_ws;
    unsigned short* Xb  = p;           p += nX;
    unsigned short* Wqb = p;           p += nWq;
    unsigned short* Wkb = p;           p += nWq;
    unsigned short* Wvb = p;           p += nWv;
    unsigned short* Wob = p;           p += nWv;
    unsigned short* Qp  = p;           p += nQp;
    unsigned short* Kp  = p;           p += nQp;
    unsigned short* Vt  = p;           p += nVt;
    unsigned short* Cx  = p;           p += nCx;

    const int B = 256;
    conv_f32_bf16<<<(int)((nX  + B - 1) / B), B, 0, stream>>>(X,  Xb,  (int)nX);
    conv_f32_bf16<<<(int)((nWq + B - 1) / B), B, 0, stream>>>(Wq, Wqb, (int)nWq);
    conv_f32_bf16<<<(int)((nWq + B - 1) / B), B, 0, stream>>>(Wk, Wkb, (int)nWq);
    conv_f32_bf16<<<(int)((nWv + B - 1) / B), B, 0, stream>>>(Wv, Wvb, (int)nWv);
    conv_f32_bf16<<<(int)((nWv + B - 1) / B), B, 0, stream>>>(Wo, Wob, (int)nWv);

    // QKV projections (fused per-head repack in the store path)
    gemm_nt_bf16<<<dim3(Sq / 128, NQ / 64), 256, 0, stream>>>(
        Xb, Wqb, nullptr, Qp, Sq, NQ, HID, 1);
    gemm_nt_bf16<<<dim3(Sq / 128, NQ / 64), 256, 0, stream>>>(
        Xb, Wkb, nullptr, Kp, Sq, NQ, HID, 1);
    gemm_nt_bf16<<<dim3(Sq / 128, NV / 64), 256, 0, stream>>>(
        Xb, Wvb, nullptr, Vt, Sq, NV, HID, 2);

    // RoPE; fold 1/sqrt(DN), 1/sqrt(DR) into Q only
    rope_scale<<<(Hh * Sq) / 8, 256, 0, stream>>>(Qp, 0.088388347648f /*1/sqrt(128)*/,
                                                  0.125f /*1/sqrt(64)*/);
    rope_scale<<<(Hh * Sq) / 8, 256, 0, stream>>>(Kp, 1.0f, 1.0f);

    // attention
    flash_attn<<<dim3(Hh, Sq / 128), 256, 0, stream>>>(Qp, Kp, Vt, Cx);

    // output projection -> f32
    gemm_nt_bf16<<<dim3(Sq / 128, HID / 64), 256, 0, stream>>>(
        Cx, Wob, out, nullptr, Sq, HID, HID, 0);
}
